// SwinWindowAttention_36996848288290
// MI455X (gfx1250) — matrix operand order repaired and checked by reading, not compiled, not run on hardware
//
#include <hip/hip_runtime.h>

// ---------------------------------------------------------------------------
// Fused Swin window attention for gfx1250 (MI455X), round 6.
// One workgroup per (batch,window); whole block resident in ~276 KB LDS.
// All GEMMs via v_wmma_f32_16x16x32_bf16 (wave32).
//  - phase-2 QK: 3 column-tiles per wave in flight -> A fragments loaded once
//  - fragment loads batched ahead of each WMMA chain (partial dscnt waits)
//  - bias table staged via async global->LDS loads (global_load_async_to_lds)
//  - attention output kept in bf16 (it only feeds a bf16 WMMA A-operand)
// ---------------------------------------------------------------------------

typedef __bf16 v16bf __attribute__((ext_vector_type(16)));
typedef __bf16 v8bf  __attribute__((ext_vector_type(8)));
typedef float  v8f   __attribute__((ext_vector_type(8)));

constexpr int WAVES  = 8;
constexpr int BLOCK  = WAVES * 32;

constexpr int N_TOK  = 98;          // 2*7*7 tokens per window
constexpr int NP     = 112;         // padded to 7 tiles of 16
constexpr int C_DIM  = 192;
constexpr int H_HEADS= 6;
constexpr int HD     = 32;
constexpr int NW     = 64;
constexpr int QKV_COLS = 3 * C_DIM; // 576
constexpr int MT     = 7;           // M tiles (rows)
constexpr int TBL    = 507 * H_HEADS;

// LDS strides (elements). Row-byte-stride % 16 == 0 (b128 alignment) and
// row stride mod 64 banks != 0 (conflict-free column walks).
constexpr int XLD    = 200;  // bf16, x staging            (400 B/row)
constexpr int QKLD   = 392;  // bf16, Q|K (cols 0..383)    (784 B/row)
constexpr int VTLD   = 136;  // bf16, V transposed [192][] (272 B/row)
constexpr int OUTLD  = 200;  // bf16, attention output     (400 B/row)
constexpr int PLD    = 136;  // bf16, per-wave P tile      (272 B/row)

__device__ __forceinline__ v8f wmma_bf16(v16bf a, v16bf b, v8f c) {
  return __builtin_amdgcn_wmma_f32_16x16x32_bf16(
      false, a, false, b, (short)0, c, false, false);
}

// ---- fragment loaders (contiguous-k layouts, wave32), all unconditional ----

// A 16x32 bf16 from row-major LDS: lane L holds row M=L%16; per lane the 16
// elems are one contiguous 16-elem run at k0+16*(L/16).
__device__ __forceinline__ v16bf load_a_bf(const __bf16* p, int ld, int m0, int k0) {
  int lane = threadIdx.x & 31;
  const __bf16* q = p + (m0 + (lane & 15)) * ld + k0 + ((lane >> 4) << 4);
  v8bf lo = *(const v8bf*)q;
  v8bf hi = *(const v8bf*)(q + 8);
  return __builtin_shufflevector(lo, hi, 0, 1, 2, 3, 4, 5, 6, 7,
                                 8, 9, 10, 11, 12, 13, 14, 15);
}

// B 32x16 bf16 where the source is stored n-major (element (k,n)=p[n*ld+k]):
// lane L holds col N=L%16; its 16 k-values are one contiguous 16-elem run.
__device__ __forceinline__ v16bf load_b_cm(const __bf16* p, int ld, int k0, int n0) {
  int lane = threadIdx.x & 31;
  const __bf16* q = p + (n0 + (lane & 15)) * ld + k0 + ((lane >> 4) << 4);
  v8bf lo = *(const v8bf*)q;
  v8bf hi = *(const v8bf*)(q + 8);
  return __builtin_shufflevector(lo, hi, 0, 1, 2, 3, 4, 5, 6, 7,
                                 8, 9, 10, 11, 12, 13, 14, 15);
}

// B 32x16 from global fp32 weights, row-major (k,n)=p[k*ld+n]. Statically
// in-bounds -> branchless; k-strided dword loads, L2-resident weights.
__device__ __forceinline__ v16bf load_b_gw(const float* __restrict__ p, int ld,
                                           int k0, int n0) {
  int lane = threadIdx.x & 31;
  int n = n0 + (lane & 15);
  const float* q = p + (k0 + ((lane >> 4) << 4)) * ld + n;
  v16bf t;
#pragma unroll
  for (int e = 0; e < 16; ++e) t[e] = (__bf16)q[e * ld];
  return t;
}

// ---------------------------------------------------------------------------
__global__ __launch_bounds__(BLOCK)
__attribute__((amdgpu_waves_per_eu(2)))
void swin_attn_kernel(
    const float* __restrict__ x,       // [B, 98, 192]
    const float* __restrict__ mask,    // [64, 98, 98]
    const float* __restrict__ qkv_w,   // [192, 576]
    const float* __restrict__ qkv_b,   // [576]
    const float* __restrict__ proj_w,  // [192, 192]
    const float* __restrict__ proj_b,  // [192]
    const float* __restrict__ btable,  // [507, 6]
    float* __restrict__ out)           // [B, 98, 192]
{
  __shared__ __bf16 xbuf[NP * XLD];               // 44,800 B
  __shared__ __bf16 qkbuf[NP * QKLD];             // 87,808 B (Q 0..191 | K 192..383)
  __shared__ __bf16 vtbuf[C_DIM * VTLD];          // 52,224 B (V^T [feature][token])
  __shared__ __bf16 outbuf[NP * OUTLD];           // 44,800 B (attention out, bf16)
  __shared__ __bf16 pbuf[WAVES * 16 * PLD];       // 34,816 B
  __shared__ float  sbtable[TBL];                 // 12,168 B -> total 276,616 B
  const int b     = blockIdx.x;
  const int tid   = threadIdx.x;
  const int lane  = tid & 31;
  const int wave  = tid >> 5;
  const int ncol0 = lane & 15;
  const int rb8   = (lane >> 4) << 3;   // 0 or 8: row offset of C-frag half
  const int w     = b & (NW - 1);
  const float* xb = x + (size_t)b * N_TOK * C_DIM;

  // ---- Phase 1: stage x (b128 stores) + bias table; zero pad regions --
  for (int i = tid; i < (N_TOK * C_DIM) / 8; i += BLOCK) {
    const float4* src = (const float4*)xb + i * 2;
    float4 va = src[0], vb = src[1];
    v8bf vv;
    vv[0] = (__bf16)va.x; vv[1] = (__bf16)va.y;
    vv[2] = (__bf16)va.z; vv[3] = (__bf16)va.w;
    vv[4] = (__bf16)vb.x; vv[5] = (__bf16)vb.y;
    vv[6] = (__bf16)vb.z; vv[7] = (__bf16)vb.w;
    int r = (i * 8) / C_DIM, c0 = (i * 8) % C_DIM;
    *(v8bf*)(&xbuf[r * XLD + c0]) = vv;
  }
  // Bias table: straight fp32 global->LDS copy -> CDNA5 async-to-LDS path.
#if __has_builtin(__builtin_amdgcn_global_load_async_to_lds_b32)
  {
    __attribute__((address_space(1))) int* gsrc =
        (__attribute__((address_space(1))) int*)btable;
    __attribute__((address_space(3))) int* ldst =
        (__attribute__((address_space(3))) int*)sbtable;
    for (int i = tid; i < TBL; i += BLOCK)
      __builtin_amdgcn_global_load_async_to_lds_b32(gsrc + i, ldst + i, 0, 0);
#if __has_builtin(__builtin_amdgcn_s_wait_asynccnt)
    __builtin_amdgcn_s_wait_asynccnt(0);
#else
    asm volatile("s_wait_asynccnt 0x0" ::: "memory");
#endif
  }
#else
  for (int i = tid; i < TBL; i += BLOCK) sbtable[i] = btable[i];
#endif
  for (int i = tid; i < (NP - N_TOK) * XLD; i += BLOCK)
    xbuf[N_TOK * XLD + i] = (__bf16)0.0f;
  for (int i = tid; i < C_DIM * (VTLD - NP); i += BLOCK) {
    int r = i / (VTLD - NP), c = i % (VTLD - NP);
    vtbuf[r * VTLD + NP + c] = (__bf16)0.0f;
  }
  for (int i = tid; i < WAVES * 16 * (PLD - NP); i += BLOCK) {
    int r = i / (PLD - NP), c = i % (PLD - NP);
    pbuf[r * PLD + NP + c] = (__bf16)0.0f;
  }
  __syncthreads();

  // ---- Phase 2: qkv = x @ W_qkv + b; q pre-scaled; V stored transposed.
  const float scale = 0.17677669529663687f;  // 32^-0.5
  // 2a: Q and K columns (nt 0..23): each wave owns 3 column-tiles at once,
  // so every A fragment is loaded once and feeds 3 WMMAs.
  {
    const int nt0 = wave;  // columns nt0, nt0+8, nt0+16
    v16bf bw[3][6];
#pragma unroll
    for (int g = 0; g < 3; ++g)
#pragma unroll
      for (int kt = 0; kt < 6; ++kt)
        bw[g][kt] = load_b_gw(qkv_w, QKV_COLS, kt * 32, (nt0 + 8 * g) * 16);
    int   ncol[3];
    float biasv[3], scv[3];
#pragma unroll
    for (int g = 0; g < 3; ++g) {
      int nt   = nt0 + 8 * g;
      ncol[g]  = nt * 16 + ncol0;
      biasv[g] = qkv_b[ncol[g]];
      scv[g]   = (nt < 12) ? scale : 1.0f;
    }
    for (int mt = 0; mt < MT; ++mt) {
      v16bf af[6];
#pragma unroll
      for (int kt = 0; kt < 6; ++kt) af[kt] = load_a_bf(xbuf, XLD, mt * 16, kt * 32);
      int mbase = mt * 16 + rb8;
#pragma unroll
      for (int g = 0; g < 3; ++g) {
        v8f c = {};
#pragma unroll
        for (int kt = 0; kt < 6; ++kt) c = wmma_bf16(af[kt], bw[g][kt], c);
#pragma unroll
        for (int r = 0; r < 8; ++r)
          qkbuf[(mbase + r) * QKLD + ncol[g]] = (__bf16)((c[r] + biasv[g]) * scv[g]);
      }
    }
  }
  // 2b: V columns (nt 24..35), stored transposed with one b128 store
  for (int nt = 24 + wave; nt < 36; nt += WAVES) {
    v16bf bw[6];
#pragma unroll
    for (int kt = 0; kt < 6; ++kt)
      bw[kt] = load_b_gw(qkv_w, QKV_COLS, kt * 32, nt * 16);
    int n = nt * 16 + ncol0;
    float bias = qkv_b[n];
    for (int mt = 0; mt < MT; ++mt) {
      v16bf af[6];
#pragma unroll
      for (int kt = 0; kt < 6; ++kt) af[kt] = load_a_bf(xbuf, XLD, mt * 16, kt * 32);
      v8f c = {};
#pragma unroll
      for (int kt = 0; kt < 6; ++kt) c = wmma_bf16(af[kt], bw[kt], c);
      int mbase = mt * 16 + rb8;
      v8bf vv;
#pragma unroll
      for (int r = 0; r < 8; ++r) vv[r] = (__bf16)(c[r] + bias);
      *(v8bf*)(vtbuf + (n - 2 * C_DIM) * VTLD + mbase) = vv;
    }
  }
  __syncthreads();

  // ---- Phase 3: per-(head, 16-row block) attention, one unit per wave -
  const float* mrow0 = mask + (size_t)w * N_TOK * N_TOK;
  for (int u = wave; u < H_HEADS * MT; u += WAVES) {
    int h  = u / MT, mt = u % MT;
    int koff = C_DIM + h * HD;

    // S = q @ k^T : K = hd = 32 -> 7 tiles, all fragments batched first
    v16bf aq = load_a_bf(qkbuf, QKLD, mt * 16, h * HD);
    v16bf bk[MT];
#pragma unroll
    for (int nt = 0; nt < MT; ++nt) bk[nt] = load_b_cm(qkbuf + koff, QKLD, 0, nt * 16);
    v8f S[MT];
#pragma unroll
    for (int nt = 0; nt < MT; ++nt) {
      v8f z = {};
      S[nt] = wmma_bf16(aq, bk[nt], z);
    }

    // Branchless bias+mask epilogue: clamped indices + selects.
    // idx(m,n) = (169*jm + 13*im + km + 253) - (169*jn + 13*in + kn)
    int coefn[MT], nclamp[MT];
    bool nvalid[MT];
#pragma unroll
    for (int nt = 0; nt < MT; ++nt) {
      int n = nt * 16 + ncol0;
      nvalid[nt] = (n < N_TOK);
      int nc = nvalid[nt] ? n : (N_TOK - 1);
      nclamp[nt] = nc;
      int in_ = nc / 14, jn = (nc / 7) % 2, kn = nc % 7;
      coefn[nt] = 169 * jn + 13 * in_ + kn;
    }
    int mbase = mt * 16 + rb8;
#pragma unroll
    for (int r = 0; r < 8; ++r) {
      int m = mbase + r;
      bool mval = (m < N_TOK);
      int mc = mval ? m : (N_TOK - 1);
      int im = mc / 14, jm = (mc / 7) % 2, km = mc % 7;
      int basem = 169 * jm + 13 * im + km + 253;
      const float* mrow = mrow0 + mc * N_TOK;
#pragma unroll
      for (int nt = 0; nt < MT; ++nt) {
        float v = S[nt][r] + sbtable[(basem - coefn[nt]) * H_HEADS + h]
                           + mrow[nclamp[nt]];
        S[nt][r] = (mval && nvalid[nt]) ? v : -1e30f;
      }
      float mx = -1e30f;
#pragma unroll
      for (int nt = 0; nt < MT; ++nt) mx = fmaxf(mx, S[nt][r]);
#pragma unroll
      for (int off = 1; off < 16; off <<= 1) mx = fmaxf(mx, __shfl_xor(mx, off, 16));
      float sum = 0.0f;
#pragma unroll
      for (int nt = 0; nt < MT; ++nt) {
        float e = __expf(S[nt][r] - mx);
        S[nt][r] = e;
        sum += e;
      }
#pragma unroll
      for (int off = 1; off < 16; off <<= 1) sum += __shfl_xor(sum, off, 16);
      float inv = 1.0f / sum;
#pragma unroll
      for (int nt = 0; nt < MT; ++nt) S[nt][r] *= inv;
    }

    // P (C-layout regs) -> per-wave LDS tile (A-layout source for PV)
    __bf16* pw = pbuf + wave * 16 * PLD;
#pragma unroll
    for (int nt = 0; nt < MT; ++nt) {
      int n = nt * 16 + ncol0;
#pragma unroll
      for (int r = 0; r < 8; ++r) pw[(rb8 + r) * PLD + n] = (__bf16)S[nt][r];
    }

    // O = P @ v : M=16, N=32 (2 tiles), K=112 padded -> 4 k-steps of 32
    v16bf pa[4];
#pragma unroll
    for (int kt = 0; kt < 4; ++kt) pa[kt] = load_a_bf(pw, PLD, 0, kt * 32);
#pragma unroll
    for (int ont = 0; ont < 2; ++ont) {
      v16bf bv[4];
#pragma unroll
      for (int kt = 0; kt < 4; ++kt)
        bv[kt] = load_b_cm(vtbuf + (h * HD + ont * 16) * VTLD, VTLD, kt * 32, 0);
      v8f c = {};
#pragma unroll
      for (int kt = 0; kt < 4; ++kt) c = wmma_bf16(pa[kt], bv[kt], c);
      int n = h * HD + ont * 16 + ncol0;
#pragma unroll
      for (int r = 0; r < 8; ++r)
        outbuf[(mbase + r) * OUTLD + n] = (__bf16)c[r];  // pad rows: finite
    }
  }
  __syncthreads();

  // ---- Phase 4: y = out @ W_proj + b_proj -> HBM ----------------------
  float* yb = out + (size_t)b * N_TOK * C_DIM;
  for (int nt = wave; nt < 12; nt += WAVES) {
    v16bf bw[6];
#pragma unroll
    for (int kt = 0; kt < 6; ++kt)
      bw[kt] = load_b_gw(proj_w, C_DIM, kt * 32, nt * 16);
    int n = nt * 16 + ncol0;
    float pb = proj_b[n];
    for (int mt = 0; mt < MT; ++mt) {
      v16bf af[6];
#pragma unroll
      for (int kt = 0; kt < 6; ++kt) af[kt] = load_a_bf(outbuf, OUTLD, mt * 16, kt * 32);
      v8f c = {};
#pragma unroll
      for (int kt = 0; kt < 6; ++kt) c = wmma_bf16(af[kt], bw[kt], c);
      int mbase = mt * 16 + rb8;
#pragma unroll
      for (int r = 0; r < 8; ++r) {
        int m = mbase + r;
        if (m < N_TOK) yb[m * C_DIM + n] = c[r] + pb;  // only guarded store
      }
    }
  }
}

extern "C" void kernel_launch(void* const* d_in, const int* in_sizes, int n_in,
                              void* d_out, int out_size, void* d_ws, size_t ws_size,
                              hipStream_t stream) {
  const float* x      = (const float*)d_in[0];
  const float* mask   = (const float*)d_in[1];
  const float* qkv_w  = (const float*)d_in[2];
  const float* qkv_b  = (const float*)d_in[3];
  const float* proj_w = (const float*)d_in[4];
  const float* proj_b = (const float*)d_in[5];
  const float* btable = (const float*)d_in[6];
  float* out = (float*)d_out;

  int B = in_sizes[0] / (N_TOK * C_DIM);  // 2048
  swin_attn_kernel<<<B, BLOCK, 0, stream>>>(x, mask, qkv_w, qkv_b,
                                            proj_w, proj_b, btable, out);
}
